// AttentionBlock_89421219103039
// MI455X (gfx1250) — matrix operand order repaired and checked by reading, not compiled, not run on hardware
//
#include <hip/hip_runtime.h>
#include <hip/hip_bf16.h>

typedef __attribute__((ext_vector_type(16))) _Float16 v16h;
typedef __attribute__((ext_vector_type(8)))  _Float16 v8h;
typedef __attribute__((ext_vector_type(8)))  float    v8f;
typedef int i32x4 __attribute__((vector_size(16)));

#ifndef __has_builtin
#define __has_builtin(x) 0
#endif
#if __has_builtin(__builtin_amdgcn_global_load_async_to_lds_b128)
#define HAVE_ASYNC 1
#else
#define HAVE_ASYNC 0
#endif

#define AS1 __attribute__((address_space(1)))
#define AS3 __attribute__((address_space(3)))

// ---------------------------------------------------------------------------
// Async global -> LDS copy (CDNA5 GLOBAL_LOAD_ASYNC_TO_LDS_B128, ASYNCcnt).
// Fallback: plain 16B copy through VGPRs if the builtin is unavailable.
// ---------------------------------------------------------------------------
#if HAVE_ASYNC
__device__ __forceinline__ void cp_b128(void* l, const void* g) {
  __builtin_amdgcn_global_load_async_to_lds_b128(
      (AS1 i32x4*)g, (AS3 i32x4*)l, 0, 0);
}
#define ASYNC_WAIT(imm) asm volatile("s_wait_asynccnt " #imm ::: "memory")
#else
__device__ __forceinline__ void cp_b128(void* l, const void* g) {
  *(v8h*)l = *(const v8h*)g;
}
#define ASYNC_WAIT(imm) do {} while (0)
#endif

// ---------------------------------------------------------------------------
// WMMA helpers (gfx1250, wave32). D = A(16x32 f16) x B(32x16 f16) + C(f32)
// ---------------------------------------------------------------------------
__device__ __forceinline__ v8f wmma_f16(v16h a, v16h b, v8f c) {
  return __builtin_amdgcn_wmma_f32_16x16x32_f16(
      /*neg_a=*/false, a, /*neg_b=*/false, b,
      /*c_mod=*/(short)0, c, /*reuse_a=*/false, /*reuse_b=*/false);
}

// A fragment: element (M,K) with M = lane&15,
// K = (laneHalf?16:0) + j
__device__ __forceinline__ v16h load_A_frag(const _Float16* base, int row0, int ld) {
  int lane = threadIdx.x & 31;
  int m  = lane & 15;
  int kb = (lane & 16) ? 16 : 0;
  const _Float16* p = base + (size_t)(row0 + m) * ld;
  v8h lo = *(const v8h*)(p + kb);
  v8h hi = *(const v8h*)(p + 8 + kb);
  return __builtin_shufflevector(lo, hi, 0,1,2,3,4,5,6,7,8,9,10,11,12,13,14,15);
}

// B fragment: element (K,N) with N = lane&15, K = (laneHalf?16:0) + j
// base is an LDS tile stored [Ncols][Kvals] (row-major in N) => one v16h load.
__device__ __forceinline__ v16h load_B_frag(const _Float16* base, int col0, int ld) {
  int lane = threadIdx.x & 31;
  int n  = lane & 15;
  int kb = (lane & 16) ? 16 : 0;
  return *(const v16h*)(base + (size_t)(col0 + n) * ld + kb);
}

__device__ __forceinline__ float redmax16(float v) {
#pragma unroll
  for (int m = 1; m < 16; m <<= 1) v = fmaxf(v, __shfl_xor(v, m, 32));
  return v;
}
__device__ __forceinline__ float redsum16(float v) {
#pragma unroll
  for (int m = 1; m < 16; m <<= 1) v += __shfl_xor(v, m, 32);
  return v;
}

// ---------------------------------------------------------------------------
// Problem constants
// ---------------------------------------------------------------------------
#define BATCH 16
#define CHN   512
#define TLEN  1024
#define HEADS 8
#define CH    64      // per-head channels
#define GRP   32
#define CPG   16      // channels per group

// ---------------------------------------------------------------------------
// Kernel 0: f32 -> f16 weight conversion
// ---------------------------------------------------------------------------
__global__ void cvt_weights_kernel(const float* __restrict__ qw,
                                   const float* __restrict__ pw,
                                   _Float16* __restrict__ qwh,
                                   _Float16* __restrict__ pwh) {
  int i = blockIdx.x * 256 + threadIdx.x;
  if (i < 3 * CHN * CHN) qwh[i] = (_Float16)qw[i];
  if (i < CHN * CHN)     pwh[i] = (_Float16)pw[i];
}

// ---------------------------------------------------------------------------
// Kernel 1: GroupNorm, output f16 transposed to [b][t][c] for GEMM B-tiles
// ---------------------------------------------------------------------------
__global__ void groupnorm_kernel(const float* __restrict__ x,
                                 const float* __restrict__ nw,
                                 const float* __restrict__ nb,
                                 _Float16* __restrict__ hn_t) {
  __shared__ float ssum[256];
  __shared__ float ssq[256];
  int g = blockIdx.x & 31, b = blockIdx.x >> 5;
  int tid = threadIdx.x;
  const float* xg = x + ((size_t)b * CHN + g * CPG) * TLEN;

  float s = 0.f, sq = 0.f;
  for (int i = tid; i < CPG * TLEN; i += 256) {
    float v = xg[i]; s += v; sq += v * v;
  }
  ssum[tid] = s; ssq[tid] = sq;
  __syncthreads();
  for (int off = 128; off > 0; off >>= 1) {
    if (tid < off) { ssum[tid] += ssum[tid + off]; ssq[tid] += ssq[tid + off]; }
    __syncthreads();
  }
  const float inv_n = 1.f / (float)(CPG * TLEN);
  float mu  = ssum[0] * inv_n;
  float var = ssq[0] * inv_n - mu * mu;
  float rstd = rsqrtf(var + 1e-5f);

  for (int i = tid; i < CPG * TLEN; i += 256) {
    int cl = i >> 10, tt = i & 1023;
    int c = g * CPG + cl;
    float v = (xg[i] - mu) * rstd * nw[c] + nb[c];
    hn_t[((size_t)b * TLEN + tt) * CHN + c] = (_Float16)v;
  }
}

// ---------------------------------------------------------------------------
// Kernel 2: QKV GEMM  out[o,t] = sum_c W[o,c]*hn[b,c,t] + bias
// grid (24 Mtiles, 8 Ntiles, 16 b), 256 thr (8 waves, 2x4 wave grid),
// workgroup tile 64(M) x 128(N), wave tile 32x32 (2x2 WMMA frags).
// Double-buffered async-to-LDS staging (3 x b128 per thread per K-tile).
// Epilogue scatters q,k (scaled) -> [bh][t][ch], v -> [bh][ch][t], all f16.
// ---------------------------------------------------------------------------
__global__ void gemm_qkv_kernel(const _Float16* __restrict__ w_h,   // [1536][512]
                                const _Float16* __restrict__ hn_t,  // [b][1024][512]
                                const float* __restrict__ qkv_b,
                                _Float16* __restrict__ q_t,
                                _Float16* __restrict__ k_t,
                                _Float16* __restrict__ v_ch) {
  __shared__ __align__(64) _Float16 lds_a[2][64 * 32];
  __shared__ __align__(64) _Float16 lds_b[2][128 * 32];

  int tid = threadIdx.x;
  int wave = tid >> 5, lane = tid & 31;
  int wm = wave >> 2, wn = wave & 3;
  int o0 = blockIdx.x * 64, t0 = blockIdx.y * 128, bb = blockIdx.z;
  const _Float16* Bsrc = hn_t + ((size_t)bb * TLEN + t0) * CHN;

  // per-thread 16B chunk coordinates (A: 64x32 halves, B: 128x32 halves)
  int ar  = tid >> 2, ac = (tid & 3) * 8;   // A: 1 chunk
  int br0 = tid >> 2, bc = (tid & 3) * 8;   // B: 2 chunks (rows br0, br0+64)
  int br1 = br0 + 64;

  auto issue_tile = [&](int buf, int k0) {
    cp_b128(&lds_a[buf][ar  * 32 + ac], w_h  + (size_t)(o0 + ar) * CHN + k0 + ac);
    cp_b128(&lds_b[buf][br0 * 32 + bc], Bsrc + (size_t)br0 * CHN + k0 + bc);
    cp_b128(&lds_b[buf][br1 * 32 + bc], Bsrc + (size_t)br1 * CHN + k0 + bc);
  };

  v8f acc[2][2] = {};
  issue_tile(0, 0);
  for (int kk = 0; kk < CHN / 32; ++kk) {
    if (kk < CHN / 32 - 1) {
      issue_tile((kk + 1) & 1, (kk + 1) * 32);
      ASYNC_WAIT(0x3);          // current tile landed; next tile in flight
    } else {
      ASYNC_WAIT(0x0);
    }
    __syncthreads();
    const _Float16* la = lds_a[kk & 1];
    const _Float16* lb = lds_b[kk & 1];
    v16h a0 = load_A_frag(la, wm * 32,      32);
    v16h a1 = load_A_frag(la, wm * 32 + 16, 32);
    v16h b0 = load_B_frag(lb, wn * 32,      32);
    v16h b1 = load_B_frag(lb, wn * 32 + 16, 32);
    acc[0][0] = wmma_f16(a0, b0, acc[0][0]);
    acc[0][1] = wmma_f16(a0, b1, acc[0][1]);
    acc[1][0] = wmma_f16(a1, b0, acc[1][0]);
    acc[1][1] = wmma_f16(a1, b1, acc[1][1]);
    __syncthreads();            // protect buffer before next overwrite
  }

  const float scale = 0.35355339059327373f;  // ch^-1/4, applied to q and k
  int half8 = (lane & 16) ? 8 : 0;
#pragma unroll
  for (int sm = 0; sm < 2; ++sm)
#pragma unroll
    for (int sn = 0; sn < 2; ++sn)
#pragma unroll
      for (int r = 0; r < 8; ++r) {
        int o  = o0 + wm * 32 + sm * 16 + r + half8;
        int tt = t0 + wn * 32 + sn * 16 + (lane & 15);
        float val = acc[sm][sn][r] + qkv_b[o];
        int head = o / (3 * CH), rr = o % (3 * CH);
        size_t bh = (size_t)bb * HEADS + head;
        if (rr < CH)
          q_t[(bh * TLEN + tt) * CH + rr] = (_Float16)(val * scale);
        else if (rr < 2 * CH)
          k_t[(bh * TLEN + tt) * CH + (rr - CH)] = (_Float16)(val * scale);
        else
          v_ch[(bh * CH + (rr - 2 * CH)) * TLEN + tt] = (_Float16)val;
      }
}

// ---------------------------------------------------------------------------
// Kernel 3: flash attention. grid (16 qtiles, 128 bh), 128 thr (4 waves).
// Each wave owns 16 query rows: S = Q Kᵀ (2 K-steps), online softmax,
// P staged via LDS (D-layout -> A-layout), O += P V.
// K/V tiles double-buffered via async-to-LDS (8 x b128 per thread per tile).
// ---------------------------------------------------------------------------
__global__ void flash_attn_kernel(const _Float16* __restrict__ q_t,   // [bh][t][ch]
                                  const _Float16* __restrict__ k_t,   // [bh][t][ch]
                                  const _Float16* __restrict__ v_ch,  // [bh][ch][t]
                                  _Float16* __restrict__ a_t) {       // [b][t][c]
  __shared__ __align__(64) _Float16 lds_q[64 * 64];
  __shared__ __align__(64) _Float16 lds_k[2][64 * 64];
  __shared__ __align__(64) _Float16 lds_v[2][64 * 64];
  __shared__ __align__(64) _Float16 lds_p[4 * 16 * 64];

  int tid = threadIdx.x;                 // 128
  int wave = tid >> 5, lane = tid & 31;
  int tq0 = blockIdx.x * 64;
  int bh  = blockIdx.y;
  const _Float16* Q = q_t + ((size_t)bh * TLEN + tq0) * CH;
  const _Float16* K = k_t + (size_t)bh * TLEN * CH;
  const _Float16* V = v_ch + (size_t)bh * CH * TLEN;

  // Q tile: 64x64 halves = 512 x 16B chunks, 4 per thread
#pragma unroll
  for (int j = 0; j < 4; ++j) {
    int c = tid + j * 128;
    int r = c >> 3, ck = (c & 7) * 8;
    cp_b128(lds_q + r * 64 + ck, Q + (size_t)r * CH + ck);
  }

  auto issue_kv = [&](int buf, int kb0) {
#pragma unroll
    for (int j = 0; j < 4; ++j) {
      int c = tid + j * 128;
      int r = c >> 3, ck = (c & 7) * 8;
      cp_b128(&lds_k[buf][r * 64 + ck], K + (size_t)(kb0 + r) * CH + ck);
      cp_b128(&lds_v[buf][r * 64 + ck], V + (size_t)r * TLEN + kb0 + ck);
    }
  };

  float mrow[8], lrow[8];
  v8f oacc[4] = {};
#pragma unroll
  for (int r = 0; r < 8; ++r) { mrow[r] = -1e30f; lrow[r] = 0.f; }
  _Float16* pw = lds_p + wave * 16 * 64;
  int rb = wave * 16;
  int half8 = (lane & 16) ? 8 : 0;

  issue_kv(0, 0);
  for (int it = 0; it < TLEN / 64; ++it) {
    if (it < TLEN / 64 - 1) {
      issue_kv((it + 1) & 1, (it + 1) * 64);
      ASYNC_WAIT(0x8);          // Q + current K/V tile landed
    } else {
      ASYNC_WAIT(0x0);
    }
    __syncthreads();
    const _Float16* lk = lds_k[it & 1];
    const _Float16* lv = lds_v[it & 1];

    v16h aq0 = load_A_frag(lds_q,      rb, 64);   // K (=ch) 0..31
    v16h aq1 = load_A_frag(lds_q + 32, rb, 64);   // K (=ch) 32..63
    v8f s[4];
#pragma unroll
    for (int f = 0; f < 4; ++f) {
      v8f z = {};
      z = wmma_f16(aq0, load_B_frag(lk,      f * 16, 64), z);
      z = wmma_f16(aq1, load_B_frag(lk + 32, f * 16, 64), z);
      s[f] = z;
    }

#pragma unroll
    for (int r = 0; r < 8; ++r) {
      float mx = fmaxf(fmaxf(s[0][r], s[1][r]), fmaxf(s[2][r], s[3][r]));
      mx = redmax16(mx);
      float mnew  = fmaxf(mrow[r], mx);
      float alpha = __expf(mrow[r] - mnew);
      float rsum = 0.f;
#pragma unroll
      for (int f = 0; f < 4; ++f) {
        float p = __expf(s[f][r] - mnew);
        rsum += p;
        pw[(r + half8) * 64 + f * 16 + (lane & 15)] = (_Float16)p;
      }
      rsum = redsum16(rsum);
      lrow[r] = lrow[r] * alpha + rsum;
      mrow[r] = mnew;
#pragma unroll
      for (int n = 0; n < 4; ++n) oacc[n][r] *= alpha;
    }
    asm volatile("s_wait_dscnt 0" ::: "memory");   // P store -> P A-frag load

    v16h ap0 = load_A_frag(pw,      0, 64);        // key 0..31
    v16h ap1 = load_A_frag(pw + 32, 0, 64);        // key 32..63
#pragma unroll
    for (int n = 0; n < 4; ++n) {
      oacc[n] = wmma_f16(ap0, load_B_frag(lv,      n * 16, 64), oacc[n]);
      oacc[n] = wmma_f16(ap1, load_B_frag(lv + 32, n * 16, 64), oacc[n]);
    }
    __syncthreads();            // protect K/V buffer before next overwrite
  }

  int b = bh >> 3, head = bh & 7;
#pragma unroll
  for (int n = 0; n < 4; ++n)
#pragma unroll
    for (int r = 0; r < 8; ++r) {
      int row = tq0 + rb + r + half8;
      int col = head * CH + n * 16 + (lane & 15);
      float val = oacc[n][r] / lrow[r];
      a_t[((size_t)b * TLEN + row) * CHN + col] = (_Float16)val;
    }
}

// ---------------------------------------------------------------------------
// Kernel 4: proj GEMM + bias + residual, writes final f32 output [b][c][t]
// ---------------------------------------------------------------------------
__global__ void gemm_proj_kernel(const _Float16* __restrict__ w_h,  // [512][512]
                                 const _Float16* __restrict__ a_t,  // [b][1024][512]
                                 const float* __restrict__ pb,
                                 const float* __restrict__ x,
                                 float* __restrict__ out) {
  __shared__ __align__(64) _Float16 lds_a[2][64 * 32];
  __shared__ __align__(64) _Float16 lds_b[2][128 * 32];

  int tid = threadIdx.x;
  int wave = tid >> 5, lane = tid & 31;
  int wm = wave >> 2, wn = wave & 3;
  int o0 = blockIdx.x * 64, t0 = blockIdx.y * 128, bb = blockIdx.z;
  const _Float16* Bsrc = a_t + ((size_t)bb * TLEN + t0) * CHN;

  int ar  = tid >> 2, ac = (tid & 3) * 8;
  int br0 = tid >> 2, bc = (tid & 3) * 8;
  int br1 = br0 + 64;

  auto issue_tile = [&](int buf, int k0) {
    cp_b128(&lds_a[buf][ar  * 32 + ac], w_h  + (size_t)(o0 + ar) * CHN + k0 + ac);
    cp_b128(&lds_b[buf][br0 * 32 + bc], Bsrc + (size_t)br0 * CHN + k0 + bc);
    cp_b128(&lds_b[buf][br1 * 32 + bc], Bsrc + (size_t)br1 * CHN + k0 + bc);
  };

  v8f acc[2][2] = {};
  issue_tile(0, 0);
  for (int kk = 0; kk < CHN / 32; ++kk) {
    if (kk < CHN / 32 - 1) {
      issue_tile((kk + 1) & 1, (kk + 1) * 32);
      ASYNC_WAIT(0x3);
    } else {
      ASYNC_WAIT(0x0);
    }
    __syncthreads();
    const _Float16* la = lds_a[kk & 1];
    const _Float16* lb = lds_b[kk & 1];
    v16h a0 = load_A_frag(la, wm * 32,      32);
    v16h a1 = load_A_frag(la, wm * 32 + 16, 32);
    v16h b0 = load_B_frag(lb, wn * 32,      32);
    v16h b1 = load_B_frag(lb, wn * 32 + 16, 32);
    acc[0][0] = wmma_f16(a0, b0, acc[0][0]);
    acc[0][1] = wmma_f16(a0, b1, acc[0][1]);
    acc[1][0] = wmma_f16(a1, b0, acc[1][0]);
    acc[1][1] = wmma_f16(a1, b1, acc[1][1]);
    __syncthreads();
  }

  int half8 = (lane & 16) ? 8 : 0;
#pragma unroll
  for (int sm = 0; sm < 2; ++sm)
#pragma unroll
    for (int sn = 0; sn < 2; ++sn)
#pragma unroll
      for (int r = 0; r < 8; ++r) {
        int o  = o0 + wm * 32 + sm * 16 + r + half8;
        int tt = t0 + wn * 32 + sn * 16 + (lane & 15);
        size_t idx = ((size_t)bb * CHN + o) * TLEN + tt;
        out[idx] = acc[sm][sn][r] + pb[o] + x[idx];
      }
}

// ---------------------------------------------------------------------------
// Launch
// ---------------------------------------------------------------------------
extern "C" void kernel_launch(void* const* d_in, const int* in_sizes, int n_in,
                              void* d_out, int out_size, void* d_ws, size_t ws_size,
                              hipStream_t stream) {
  const float* x      = (const float*)d_in[0];
  const float* norm_w = (const float*)d_in[1];
  const float* norm_b = (const float*)d_in[2];
  const float* qkv_w  = (const float*)d_in[3];
  const float* qkv_b  = (const float*)d_in[4];
  const float* proj_w = (const float*)d_in[5];
  const float* proj_b = (const float*)d_in[6];
  float* out = (float*)d_out;

  char* ws = (char*)d_ws;
  // workspace layout (bytes)
  _Float16* qwh  = (_Float16*)(ws + 0);           //  1,572,864  qkv_w f16
  _Float16* pwh  = (_Float16*)(ws + 1572864);     //    524,288  proj_w f16
  _Float16* hn_t = (_Float16*)(ws + 2097152);     // 16,777,216  hn [b][t][c]
  _Float16* q_t  = (_Float16*)(ws + 18874368);    // 16,777,216  q  [bh][t][ch]
  _Float16* k_t  = (_Float16*)(ws + 35651584);    // 16,777,216  k  [bh][t][ch]
  _Float16* v_ch = (_Float16*)(ws + 52428800);    // 16,777,216  v  [bh][ch][t]
  _Float16* a_t  = (_Float16*)(ws + 69206016);    // 16,777,216  a  [b][t][c]

  cvt_weights_kernel<<<(3 * CHN * CHN + 255) / 256, 256, 0, stream>>>(qkv_w, proj_w, qwh, pwh);
  groupnorm_kernel<<<BATCH * GRP, 256, 0, stream>>>(x, norm_w, norm_b, hn_t);
  dim3 gq(3 * CHN / 64, TLEN / 128, BATCH);
  gemm_qkv_kernel<<<gq, 256, 0, stream>>>(qwh, hn_t, qkv_b, q_t, k_t, v_ch);
  dim3 gf(TLEN / 64, BATCH * HEADS);
  flash_attn_kernel<<<gf, 128, 0, stream>>>(q_t, k_t, v_ch, a_t);
  dim3 gp(CHN / 64, TLEN / 128, BATCH);
  gemm_proj_kernel<<<gp, 256, 0, stream>>>(pwh, a_t, proj_b, x, out);
}
